// Int8Linear_72962904424564
// MI455X (gfx1250) — matrix operand order repaired and checked by reading, not compiled, not run on hardware
//
#include <hip/hip_runtime.h>

// ---------------------------------------------------------------------------
// int8 block-dequant linear:  out = x @ dequant(w_q, w_s).T + bias
// M=8192 K=4096 N=16384.  AI ~= 1170 flop/byte -> compute-bound -> the MI455X
// answer is bf16 WMMA fed by the Tensor Data Mover:
//   pass 1: x f32 -> bf16, w_q int32 * blockscale -> bf16   (workspace)
//   pass 2: bf16 GEMM, 128x128 C tile / WG, BK=64, double-buffered LDS
//           staged by TENSOR_LOAD_TO_LDS (TENSORcnt), 16 wmma / wave / stage
// Fallback (small ws): fused single-pass WMMA kernel; no TDM builtin -> plain LDS copy.
// ---------------------------------------------------------------------------

typedef __attribute__((ext_vector_type(16))) __bf16 v16bf;
typedef __attribute__((ext_vector_type(8)))  __bf16 bf16x8;
typedef __attribute__((ext_vector_type(8)))  float  v8f;
typedef __attribute__((ext_vector_type(4)))  float  f32x4;
typedef __attribute__((ext_vector_type(4)))  int    i32x4;
typedef __attribute__((ext_vector_type(8)))  int    i32x8;
typedef __attribute__((ext_vector_type(4)))  unsigned int u32x4;

#if defined(__has_builtin)
#if __has_builtin(__builtin_amdgcn_tensor_load_to_lds) && \
    __has_builtin(__builtin_amdgcn_s_wait_tensorcnt)
#define HAVE_TDM 1
#endif
#endif

// ======================= TDM descriptor helper ==============================
// 2D tensor load: tile [tile_d1 rows x tile_d0 elems], 2-byte elements,
// row stride stride0 (elems).  LDS padding: +4 DWORDs every 32 DWORDs
// (pad_interval code 4, pad_amount code 3) -> LDS row stride = 144B.
#ifdef HAVE_TDM
__device__ __forceinline__ void tdm_load_2d_bf16(
    unsigned int lds_byte_addr, const void* gptr,
    unsigned int tensor_d0, unsigned int tensor_d1,
    unsigned int tile_d0, unsigned int tile_d1,
    unsigned long long stride0_elems)
{
    unsigned long long ga = (unsigned long long)gptr;
    u32x4 g0;
    g0[0] = 1u;                                   // count=1, user descriptor
    g0[1] = lds_byte_addr;                        // LDS byte address
    g0[2] = (unsigned int)(ga & 0xFFFFFFFFu);     // global_addr[31:0]
    g0[3] = (unsigned int)((ga >> 32) & 0x01FFFFFFu) | (2u << 30); // type=2
    i32x8 g1;
    // data_size=1(2B) | pad_enable | pad_interval=4(32dw) | pad_amount=3(4dw)
    g1[0] = (int)((1u << 16) | (1u << 20) | (4u << 22) | (3u << 25));
    g1[1] = (int)((tensor_d0 & 0xFFFFu) << 16);                 // dim0[15:0]
    g1[2] = (int)((tensor_d0 >> 16) | ((tensor_d1 & 0xFFFFu) << 16));
    g1[3] = (int)((tensor_d1 >> 16) | (tile_d0 << 16));
    g1[4] = (int)(tile_d1 & 0xFFFFu);                           // tile_dim2=0
    g1[5] = (int)(unsigned int)(stride0_elems & 0xFFFFFFFFull);
    g1[6] = (int)(unsigned int)((stride0_elems >> 32) & 0xFFFFull);
    g1[7] = 0;
    i32x4 z4 = {0, 0, 0, 0};
#if defined(__clang_major__) && (__clang_major__ >= 23)
    i32x8 z8 = {0, 0, 0, 0, 0, 0, 0, 0};
    __builtin_amdgcn_tensor_load_to_lds(g0, g1, z4, z4, z8, 0);
#else
    __builtin_amdgcn_tensor_load_to_lds(g0, g1, z4, z4, 0);
#endif
}
#endif

// ============================ prep kernels ==================================
__global__ __launch_bounds__(256) void cvt_x_bf16(
    const float* __restrict__ x, __bf16* __restrict__ outx, size_t n8)
{
    for (size_t i = (size_t)blockIdx.x * 256 + threadIdx.x; i < n8;
         i += (size_t)gridDim.x * 256) {
        const float* p = x + i * 8;
        f32x4 a = *(const f32x4*)p;
        f32x4 b = *(const f32x4*)(p + 4);
        bf16x8 o;
#pragma unroll
        for (int e = 0; e < 4; ++e) {
            o[e]     = (__bf16)a[e];
            o[e + 4] = (__bf16)b[e];
        }
        *(bf16x8*)(outx + i * 8) = o;
    }
}

__global__ __launch_bounds__(256) void cvt_w_bf16(
    const int* __restrict__ w_q, const float* __restrict__ w_s,
    __bf16* __restrict__ outw, size_t n8, int K)
{
    const int k8_per_row = K >> 3;
    const int kblocks    = K >> 7;
    for (size_t i = (size_t)blockIdx.x * 256 + threadIdx.x; i < n8;
         i += (size_t)gridDim.x * 256) {
        const size_t n = i / (size_t)k8_per_row;
        const int    k = (int)(i - n * (size_t)k8_per_row) << 3;
        const float  s = w_s[(n >> 7) * (size_t)kblocks + (k >> 7)];
        const int* p = w_q + n * (size_t)K + k;
        i32x4 a = *(const i32x4*)p;
        i32x4 b = *(const i32x4*)(p + 4);
        bf16x8 o;
#pragma unroll
        for (int e = 0; e < 4; ++e) {
            o[e]     = (__bf16)((float)a[e] * s);
            o[e + 4] = (__bf16)((float)b[e] * s);
        }
        *(bf16x8*)(outw + n * (size_t)K + k) = o;
    }
}

// ===================== bf16 GEMM with TDM staging ===========================
#define GBM 128
#define GBN 128
#define GBK 64
#define ROWE 72                    // 64 elems + 8 pad  -> 144B row stride
#define STAGE_ELEMS (128 * ROWE)   // 9216 elems = 18432 B per tile stage
// LDS layout: [A stage0][A stage1][B stage0][B stage1]

__global__ __launch_bounds__(256) void gemm_bf16_tdm(
    const __bf16* __restrict__ A,     // [M,K] bf16, K-contiguous
    const __bf16* __restrict__ B,     // [N,K] bf16, K-contiguous
    const float* __restrict__ bias,
    float* __restrict__ out,
    int M, int K, int N)
{
    __shared__ __align__(16) __bf16 smem[4 * STAGE_ELEMS];

    const int tid  = threadIdx.x;
    const int lane = tid & 31;
    const int wave = tid >> 5;          // 0..7
    const int wm   = wave >> 1;         // 0..3 -> 32 M-rows each
    const int wn   = wave & 1;          // 0..1 -> 64 N-cols each
    const int r15  = lane & 15;
    const int kh   = lane >> 4;

    const int n0 = blockIdx.x * GBN;
    const int m0 = blockIdx.y * GBM;

    v8f acc[2][4];
#pragma unroll
    for (int i = 0; i < 2; ++i)
#pragma unroll
        for (int j = 0; j < 4; ++j)
            acc[i][j] = (v8f){0.f,0.f,0.f,0.f,0.f,0.f,0.f,0.f};

    const int ksteps = K / GBK;

#ifdef HAVE_TDM
    const unsigned int ldsA[2] = {
        (unsigned int)(unsigned long long)&smem[0],
        (unsigned int)(unsigned long long)&smem[STAGE_ELEMS] };
    const unsigned int ldsB[2] = {
        (unsigned int)(unsigned long long)&smem[2 * STAGE_ELEMS],
        (unsigned int)(unsigned long long)&smem[3 * STAGE_ELEMS] };

    // prologue: stage 0 in flight (wave0 -> A tile, wave1 -> B tile)
    if (wave == 0)
        tdm_load_2d_bf16(ldsA[0], A + (size_t)m0 * K, K, M, GBK, GBM, K);
    else if (wave == 1)
        tdm_load_2d_bf16(ldsB[0], B + (size_t)n0 * K, K, N, GBK, GBN, K);
#else
    // cooperative fallback copy: 256 thr, each copies 32 elems of a tile
    const int crow = tid >> 1;
    const int ccol = (tid & 1) * 32;
#define COPY_TILE(dstbase, src, ld)                                        \
    {                                                                      \
        const __bf16* s_ = (src) + (size_t)crow * (ld) + ccol;             \
        __bf16* d_ = &smem[(dstbase) + crow * ROWE + ccol];                \
        _Pragma("unroll")                                                  \
        for (int j_ = 0; j_ < 4; ++j_)                                     \
            *(bf16x8*)(d_ + 8 * j_) = *(const bf16x8*)(s_ + 8 * j_);       \
    }
    COPY_TILE(0,               A + (size_t)m0 * K, K)
    COPY_TILE(2 * STAGE_ELEMS, B + (size_t)n0 * K, K)
#endif

    for (int kt = 0; kt < ksteps; ++kt) {
        const int cur = kt & 1;
#ifdef HAVE_TDM
        __builtin_amdgcn_s_wait_tensorcnt(0);   // issuing waves drain TENSORcnt
#endif
        __syncthreads();                        // stage `cur` published

        if (kt + 1 < ksteps) {
            const size_t koff = (size_t)(kt + 1) * GBK;
#ifdef HAVE_TDM
            if (wave == 0)
                tdm_load_2d_bf16(ldsA[cur ^ 1], A + (size_t)m0 * K + koff,
                                 K, M, GBK, GBM, K);
            else if (wave == 1)
                tdm_load_2d_bf16(ldsB[cur ^ 1], B + (size_t)n0 * K + koff,
                                 K, N, GBK, GBN, K);
#else
            COPY_TILE((cur ^ 1) * STAGE_ELEMS,
                      A + (size_t)m0 * K + koff, K)
            COPY_TILE((2 + (cur ^ 1)) * STAGE_ELEMS,
                      B + (size_t)n0 * K + koff, K)
#endif
        }

        const __bf16* sA = &smem[cur * STAGE_ELEMS];
        const __bf16* sB = &smem[(2 + cur) * STAGE_ELEMS];

#pragma unroll
        for (int ks = 0; ks < 2; ++ks) {        // two K=32 sub-steps
            const int kb = ks * 32;
            // A 16x32 bf16 frag: lane=(row m), regs = two 8-K runs
            v16bf af[2];
#pragma unroll
            for (int i = 0; i < 2; ++i) {
                const __bf16* p =
                    &sA[(wm * 32 + i * 16 + r15) * ROWE + kb + kh * 8];
                bf16x8 lo = *(const bf16x8*)p;
                bf16x8 hi = *(const bf16x8*)(p + 16);
                af[i] = __builtin_shufflevector(lo, hi,
                    0,1,2,3,4,5,6,7,8,9,10,11,12,13,14,15);
            }
            // B 32x16 bf16 frag: lane=(col n), regs = two 8-K runs
            v16bf bfr[4];
#pragma unroll
            for (int j = 0; j < 4; ++j) {
                const __bf16* p =
                    &sB[(wn * 64 + j * 16 + r15) * ROWE + kb + kh * 8];
                bf16x8 lo = *(const bf16x8*)p;
                bf16x8 hi = *(const bf16x8*)(p + 16);
                bfr[j] = __builtin_shufflevector(lo, hi,
                    0,1,2,3,4,5,6,7,8,9,10,11,12,13,14,15);
            }
#pragma unroll
            for (int i = 0; i < 2; ++i)
#pragma unroll
                for (int j = 0; j < 4; ++j)
                    acc[i][j] = __builtin_amdgcn_wmma_f32_16x16x32_bf16(
                        false, af[i], false, bfr[j],
                        (short)0, acc[i][j], false, false);
        }
        __syncthreads();                        // readers done with `cur`
    }

    // epilogue: C/D layout -> lane holds col n=(lane&15), reg r = row 8*kh+r
#pragma unroll
    for (int i = 0; i < 2; ++i) {
        const int mb = m0 + wm * 32 + i * 16 + kh * 8;
#pragma unroll
        for (int j = 0; j < 4; ++j) {
            const int n   = n0 + wn * 64 + j * 16 + r15;
            const float bv = bias[n];
            float* op = out + (size_t)mb * N + n;
#pragma unroll
            for (int r = 0; r < 8; ++r)
                op[(size_t)r * N] = acc[i][j][r] + bv;
        }
    }
}

// ================= fused fallback (ws-free path) ====================
#define BM 128
#define BN 128
#define BK 32
#define LDSS 40

__global__ __launch_bounds__(256) void int8_linear_wmma_fused(
    const float* __restrict__ x, const int* __restrict__ w_q,
    const float* __restrict__ w_s, const float* __restrict__ bias,
    float* __restrict__ out, int M, int K, int N)
{
    __shared__ __bf16 ldsA[BM * LDSS];
    __shared__ __bf16 ldsB[BN * LDSS];

    const int tid  = threadIdx.x;
    const int lane = tid & 31;
    const int wave = tid >> 5;
    const int wm   = wave >> 1;
    const int wn   = wave & 1;
    const int n0 = blockIdx.x * BN;
    const int m0 = blockIdx.y * BM;
    const int kblocks = K >> 7;
    const int nblk    = n0 >> 7;
    const int srow = tid >> 1;
    const int scol = (tid & 1) << 4;
    const int r15 = lane & 15;
    const int kh  = lane >> 4;

    v8f acc[2][4];
#pragma unroll
    for (int i = 0; i < 2; ++i)
#pragma unroll
        for (int j = 0; j < 4; ++j)
            acc[i][j] = (v8f){0.f,0.f,0.f,0.f,0.f,0.f,0.f,0.f};

    for (int k0 = 0; k0 < K; k0 += BK) {
        {
            const float* gp = x + (size_t)(m0 + srow) * K + (k0 + scol);
            f32x4 v0 = *(const f32x4*)(gp + 0);
            f32x4 v1 = *(const f32x4*)(gp + 4);
            f32x4 v2 = *(const f32x4*)(gp + 8);
            f32x4 v3 = *(const f32x4*)(gp + 12);
            bf16x8 p0, p1;
#pragma unroll
            for (int e = 0; e < 4; ++e) {
                p0[e] = (__bf16)v0[e]; p0[e+4] = (__bf16)v1[e];
                p1[e] = (__bf16)v2[e]; p1[e+4] = (__bf16)v3[e];
            }
            *(bf16x8*)&ldsA[srow * LDSS + scol]     = p0;
            *(bf16x8*)&ldsA[srow * LDSS + scol + 8] = p1;
            if (k0 + BK < K) __builtin_prefetch(gp + BK, 0, 0);
        }
        {
            const float s = w_s[nblk * kblocks + (k0 >> 7)];
            const int* gp = w_q + (size_t)(n0 + srow) * K + (k0 + scol);
            i32x4 q0 = *(const i32x4*)(gp + 0);
            i32x4 q1 = *(const i32x4*)(gp + 4);
            i32x4 q2 = *(const i32x4*)(gp + 8);
            i32x4 q3 = *(const i32x4*)(gp + 12);
            bf16x8 p0, p1;
#pragma unroll
            for (int e = 0; e < 4; ++e) {
                p0[e] = (__bf16)((float)q0[e] * s); p0[e+4] = (__bf16)((float)q1[e] * s);
                p1[e] = (__bf16)((float)q2[e] * s); p1[e+4] = (__bf16)((float)q3[e] * s);
            }
            *(bf16x8*)&ldsB[srow * LDSS + scol]     = p0;
            *(bf16x8*)&ldsB[srow * LDSS + scol + 8] = p1;
            if (k0 + BK < K) __builtin_prefetch(gp + BK, 0, 0);
        }
        __syncthreads();

        v16bf afrag[2];
#pragma unroll
        for (int i = 0; i < 2; ++i) {
            const __bf16* p = &ldsA[(wm * 32 + i * 16 + r15) * LDSS + kh * 8];
            bf16x8 lo = *(const bf16x8*)p;
            bf16x8 hi = *(const bf16x8*)(p + 16);
            afrag[i] = __builtin_shufflevector(lo, hi,
                0,1,2,3,4,5,6,7,8,9,10,11,12,13,14,15);
        }
        v16bf bfrag[4];
#pragma unroll
        for (int j = 0; j < 4; ++j) {
            const __bf16* p = &ldsB[(wn * 64 + j * 16 + r15) * LDSS + kh * 8];
            bf16x8 lo = *(const bf16x8*)p;
            bf16x8 hi = *(const bf16x8*)(p + 16);
            bfrag[j] = __builtin_shufflevector(lo, hi,
                0,1,2,3,4,5,6,7,8,9,10,11,12,13,14,15);
        }
#pragma unroll
        for (int i = 0; i < 2; ++i)
#pragma unroll
            for (int j = 0; j < 4; ++j)
                acc[i][j] = __builtin_amdgcn_wmma_f32_16x16x32_bf16(
                    false, afrag[i], false, bfrag[j],
                    (short)0, acc[i][j], false, false);
        __syncthreads();
    }

#pragma unroll
    for (int i = 0; i < 2; ++i) {
        const int mb = m0 + wm * 32 + i * 16 + kh * 8;
#pragma unroll
        for (int j = 0; j < 4; ++j) {
            const int n  = n0 + wn * 64 + j * 16 + r15;
            const float bv = bias[n];
            float* op = out + (size_t)mb * N + n;
#pragma unroll
            for (int r = 0; r < 8; ++r)
                op[(size_t)r * N] = acc[i][j][r] + bv;
        }
    }
}

// ============================== launcher ====================================
extern "C" void kernel_launch(void* const* d_in, const int* in_sizes, int n_in,
                              void* d_out, int out_size, void* d_ws, size_t ws_size,
                              hipStream_t stream) {
    const float* x    = (const float*)d_in[0];
    const int*   w_q  = (const int*)d_in[1];
    const float* w_s  = (const float*)d_in[2];
    const float* bias = (const float*)d_in[3];
    float*       out  = (float*)d_out;

    const int N = in_sizes[3];
    const int K = in_sizes[1] / N;
    const int M = in_sizes[0] / K;

    const size_t needA = (size_t)M * (size_t)K * 2;
    const size_t needB = (size_t)N * (size_t)K * 2;

    if (ws_size >= needA + needB) {
        __bf16* Abf = (__bf16*)d_ws;
        __bf16* Bbf = (__bf16*)((char*)d_ws + needA);
        cvt_x_bf16<<<dim3(2048), 256, 0, stream>>>(x, Abf,
                                                   (size_t)M * K / 8);
        cvt_w_bf16<<<dim3(4096), 256, 0, stream>>>(w_q, w_s, Bbf,
                                                   (size_t)N * K / 8, K);
        dim3 grid((unsigned)(N / GBN), (unsigned)(M / GBM));
        gemm_bf16_tdm<<<grid, 256, 0, stream>>>(Abf, Bbf, bias, out, M, K, N);
    } else {
        dim3 grid((unsigned)(N / BN), (unsigned)(M / BM));
        int8_linear_wmma_fused<<<grid, 256, 0, stream>>>(x, w_q, w_s, bias,
                                                         out, M, K, N);
    }
}
